// DiscMaker_66597762892278
// MI455X (gfx1250) — matrix operand order repaired and checked by reading, not compiled, not run on hardware
//
#include <hip/hip_runtime.h>
#include <cmath>

typedef __attribute__((ext_vector_type(16))) _Float16 v16h;
typedef __attribute__((ext_vector_type(8)))  _Float16 v8h;
typedef __attribute__((ext_vector_type(8)))  float    v8f;

#define Bt   256
#define Tt   512
#define INd  4
#define Mm   3
#define Dd   256
#define NSd  64
#define Hd   128
#define MDd  768
#define G3Hd 384
#define BT   16

__device__ __forceinline__ float sigm(float v) { return 1.f / (1.f + __expf(-v)); }

// A-operand fragment: lane row holds K = {base+0..15} (contiguous)
__device__ __forceinline__ v16h frag_pair(const _Float16* p) {
  v16h a;
  v8h lo = *(const v8h*)(p);
  v8h hi = *(const v8h*)(p + 8);
#pragma unroll
  for (int j = 0; j < 8; ++j) { a[j] = lo[j]; a[j + 8] = hi[j]; }
  return a;
}
// B-operand fragment pre-packed contiguously (32B aligned)
__device__ __forceinline__ v16h frag_contig(const _Float16* p) {
  return *(const v16h*)p;
}

// ---- prep: repack W_hh (128x384 f32) into f16 WMMA B-fragment order in d_ws ----
__global__ __launch_bounds__(256, 1)
void disc_prep_whh(const float* __restrict__ Whh, _Float16* __restrict__ frag) {
  int i = blockIdx.x * 256 + threadIdx.x;
  if (i >= 24 * 4 * 32 * 16) return;
  int j = i & 15, ln = (i >> 4) & 31, kk = (i >> 9) & 3, t = i >> 11;
  int k   = kk * 32 + ((ln & 16) ? 16 : 0) + j;
  int col = t * 16 + (ln & 15);
  frag[i] = (_Float16)Whh[k * G3Hd + col];
}

__global__ __launch_bounds__(256, 1)
void disc_main(const float* __restrict__ x, const float* __restrict__ y,
               const float* __restrict__ U, const float* __restrict__ S,
               const float* __restrict__ A, const float* __restrict__ Wih,
               const float* __restrict__ bih, const float* __restrict__ bhh,
               const float* __restrict__ Wd, const float* __restrict__ bd,
               const _Float16* __restrict__ whhF, float* __restrict__ out) {
  // Augmented [S;U]^T fragments: 48 tiles x 3 chunks (K=96: 64 s-dims, 4 x-dims, pad)
  __shared__ alignas(32) _Float16 sStF[73728];   // 144 KB
  __shared__ alignas(32) _Float16 sAF[49152];    // A fragments (12 tiles x 8 chunks), 96 KB
  __shared__ alignas(16) _Float16 sK[BT * MDd];  // K matrix f16, row-major [b][md]
  __shared__ alignas(16) _Float16 sHh[BT * Hd];  // h f16 [b][c]
  __shared__ alignas(16) _Float16 sSh[BT * NSd]; // s f16 [b][n]
  __shared__ alignas(16) _Float16 sXh[BT * 32];  // x_t padded to 32 cols, f16
  __shared__ float sNSt[BT * Mm * NSd];          // new_state f32 [b][m][n]
  __shared__ float sH[BT * Hd];
  __shared__ float sSt[BT * NSd];
  __shared__ float sS2U[MDd];                    // |S_md|^2 + |U_md|^2
  __shared__ float sWih[INd * G3Hd];
  __shared__ float sbih[G3Hd];
  __shared__ float sbhh[G3Hd];
  __shared__ float sWd[Hd * Mm];
  __shared__ float sbd[Mm];
  __shared__ float sy_[BT];
  __shared__ float sErr[BT];
  __shared__ unsigned sEncU[BT * Mm];
  __shared__ float sGate[BT * Mm];
  __shared__ float sGLp[BT * Mm * 4];            // gate-logit partials
  __shared__ float sTheta[BT];
  __shared__ float sb2s[BT];                     // |s_b|^2
  __shared__ float sxx[BT];                      // |x_b|^2

  const int tid = threadIdx.x;
  const int lane = tid & 31;
  const int w = tid >> 5;
  const int b0 = blockIdx.x * BT;
  const int halfoff = (lane & 16) ? 16 : 0;  // A-operand sub-row offset
  const int r15 = lane & 15;
  const int vbase = (lane & 16) ? 8 : 0;    // C/D row offset per lane half

  // ---------- one-time init ----------
  for (int i = tid; i < 73728; i += 256) {   // augmented [S;U]^T f16 fragments
    int j = i & 15;
    int f = i >> 4;
    int ln = f & 31;
    int cidx = f >> 5;          // tt*3 + kk
    int kk = cidx % 3, tt = cidx / 3;
    int md = tt * 16 + (ln & 15);
    int kl = ((ln & 16) ? 16 : 0) + j;
    float v;
    if (kk < 2) v = S[md * NSd + kk * 32 + kl];
    else        v = (kl < INd) ? U[md * INd + kl] : 0.f;
    sStF[i] = (_Float16)v;
  }
  for (int i = tid; i < 49152; i += 256) {  // A f16 fragments
    int j = i & 15, ln = (i >> 4) & 31, kk = (i >> 9) & 7, g = i >> 12;
    int m = g >> 2, nt = g & 3;
    int d = kk * 32 + ((ln & 16) ? 16 : 0) + j;
    int n = nt * 16 + (ln & 15);
    sAF[i] = (_Float16)A[(m * Dd + d) * NSd + n];
  }
  for (int md = tid; md < MDd; md += 256) {
    float acc = 0.f;
    for (int n = 0; n < NSd; ++n) { float v = S[md * NSd + n]; acc += v * v; }
    for (int ii = 0; ii < INd; ++ii) { float v = U[md * INd + ii]; acc += v * v; }
    sS2U[md] = acc;
  }
  for (int i = tid; i < INd * G3Hd; i += 256) sWih[i] = Wih[i];
  for (int i = tid; i < G3Hd; i += 256) { sbih[i] = bih[i]; sbhh[i] = bhh[i]; }
  for (int i = tid; i < Hd * Mm; i += 256) sWd[i] = (i < (Hd - 1) * Mm) ? Wd[i] : 0.f;
  if (tid < Mm) sbd[tid] = bd[tid];
  for (int i = tid; i < BT * Hd; i += 256) { sH[i] = 0.f; sHh[i] = (_Float16)0.f; }
  for (int i = tid; i < BT * NSd; i += 256) { sSt[i] = 0.f; sSh[i] = (_Float16)0.f; }
  for (int i = tid; i < BT * 32; i += 256) sXh[i] = (_Float16)0.f;  // pad stays 0
  if (tid < BT) sErr[tid] = 1.f;
  if (tid < BT * Mm) sGate[tid] = ((tid % 3) == 2) ? 0.3334f : 0.3333f;
  __syncthreads();

  for (int t = 0; t < Tt; ++t) {
    // ---------- phase 0: stream x_t, y_t; reset per-step reductions ----------
    if (tid < BT) {               // x row: f16 copy + |x|^2
      float xx = 0.f;
#pragma unroll
      for (int ii = 0; ii < INd; ++ii) {
        float v = x[((size_t)(b0 + tid) * Tt + t) * INd + ii];
        sXh[tid * 32 + ii] = (_Float16)v;
        xx += v * v;
      }
      sxx[tid] = xx;
    }
    if (tid >= 64 && tid < 64 + BT) {
      int bl = tid - 64;
      sy_[bl] = y[(size_t)(b0 + bl) * Tt + t];
    }
    if (tid >= 96 && tid < 96 + BT * Mm) sEncU[tid - 96] = 0u;
    if (tid >= 160 && tid < 160 + BT) {
      int bl = tid - 160;
      float acc = 0.f;
      for (int n = 0; n < NSd; ++n) { float v = sSt[bl * NSd + n]; acc += v * v; }
      sb2s[bl] = acc;
    }
    __syncthreads();

    // ---------- phase 1: exponent = [s,x]@[S;U]^T (WMMA) -> K, encoded ----------
    {
      const _Float16* rp = &sSh[r15 * NSd + halfoff];
      v16h a0 = frag_pair(rp);                            // K = 0..31  (s)
      v16h a1 = frag_pair(rp + 32);                       // K = 32..63 (s)
      v16h a2 = frag_pair(&sXh[r15 * 32 + halfoff]);      // K = 64..95 (x + pad)
      float b2v[8];
#pragma unroll
      for (int v = 0; v < 8; ++v) {
        int bl = v + vbase;
        b2v[v] = sb2s[bl] + sxx[bl];
      }
#pragma unroll
      for (int q = 0; q < 6; ++q) {
        int tt = w * 6 + q;
        v8f c = {};
        v16h bf0 = frag_contig(&sStF[((tt * 3 + 0) * 32 + lane) * 16]);
        v16h bf1 = frag_contig(&sStF[((tt * 3 + 1) * 32 + lane) * 16]);
        v16h bf2 = frag_contig(&sStF[((tt * 3 + 2) * 32 + lane) * 16]);
        c = __builtin_amdgcn_wmma_f32_16x16x32_f16(false, a0, false, bf0, (short)0, c, false, false);
        c = __builtin_amdgcn_wmma_f32_16x16x32_f16(false, a1, false, bf1, (short)0, c, false, false);
        c = __builtin_amdgcn_wmma_f32_16x16x32_f16(false, a2, false, bf2, (short)0, c, false, false);
        int md = tt * 16 + r15;
        float ss2 = sS2U[md];
        int m = md >> 8;
#pragma unroll
        for (int v = 0; v < 8; ++v) {
          int bl = v + vbase;
          // exponent: -0.5*(du+ds) = cross - 0.5*(|s|^2+|x|^2+|S|^2+|U|^2)
          float kv = __expf(c[v] - 0.5f * (b2v[v] + ss2));
          sK[bl * MDd + md] = (_Float16)kv;
          atomicMax(&sEncU[bl * Mm + m], __float_as_uint(kv));  // kv >= 0
        }
      }
    }
    __syncthreads();

    // ---------- phase 2: new_state = K @ A (WMMA) and gh = h @ W_hh (WMMA) ----------
    {
      int nrep = (w < 4) ? 2 : 1;
      for (int rep = 0; rep < nrep; ++rep) {
        int g = w + rep * 8;                // 12 tiles: m in 0..2, nt in 0..3
        int m = g >> 2, nt = g & 3;
        v8f c = {};
        const _Float16* rowp = &sK[r15 * MDd + m * Dd + halfoff];
#pragma unroll
        for (int kk = 0; kk < 8; ++kk) {
          v16h a = frag_pair(rowp + kk * 32);
          v16h bf = frag_contig(&sAF[((g * 8 + kk) * 32 + lane) * 16]);
          c = __builtin_amdgcn_wmma_f32_16x16x32_f16(false, a, false, bf, (short)0, c, false, false);
        }
        int n = nt * 16 + r15;
#pragma unroll
        for (int v = 0; v < 8; ++v) sNSt[((v + vbase) * Mm + m) * NSd + n] = c[v];
      }
    }
    v8f cr = {}, cz = {}, cn = {};
    {
      const _Float16* rowp = &sHh[r15 * Hd + halfoff];
#pragma unroll
      for (int kk = 0; kk < 4; ++kk) {
        v16h a = frag_pair(rowp + kk * 32);
        v16h br = frag_contig(&whhF[(((w + 0)  * 4 + kk) * 32 + lane) * 16]);
        v16h bz = frag_contig(&whhF[(((w + 8)  * 4 + kk) * 32 + lane) * 16]);
        v16h bn = frag_contig(&whhF[(((w + 16) * 4 + kk) * 32 + lane) * 16]);
        cr = __builtin_amdgcn_wmma_f32_16x16x32_f16(false, a, false, br, (short)0, cr, false, false);
        cz = __builtin_amdgcn_wmma_f32_16x16x32_f16(false, a, false, bz, (short)0, cz, false, false);
        cn = __builtin_amdgcn_wmma_f32_16x16x32_f16(false, a, false, bn, (short)0, cn, false, false);
      }
    }
    __syncthreads();  // all sHh reads done before h_new writes

    // ---------- phase 2b: GRU nonlinearity fused in registers ----------
    {
      int c0 = w * 16 + r15;  // column in [0,128) — r/z/n share it per lane
      float bir = sbih[c0], biz = sbih[128 + c0], bin_ = sbih[256 + c0];
      float bhr = sbhh[c0], bhz = sbhh[128 + c0], bhn_ = sbhh[256 + c0];
      float wir[4], wiz[4], win[4];
#pragma unroll
      for (int j = 0; j < 4; ++j) {
        wir[j] = sWih[j * G3Hd + c0];
        wiz[j] = sWih[j * G3Hd + 128 + c0];
        win[j] = sWih[j * G3Hd + 256 + c0];
      }
#pragma unroll
      for (int v = 0; v < 8; ++v) {
        int bl = v + vbase;
        float i0 = __uint_as_float(sEncU[bl * Mm + 0]);
        float i1 = __uint_as_float(sEncU[bl * Mm + 1]);
        float i2 = __uint_as_float(sEncU[bl * Mm + 2]);
        float i3 = sErr[bl];
        float gr = bir + i0 * wir[0] + i1 * wir[1] + i2 * wir[2] + i3 * wir[3];
        float gz = biz + i0 * wiz[0] + i1 * wiz[1] + i2 * wiz[2] + i3 * wiz[3];
        float gn = bin_ + i0 * win[0] + i1 * win[1] + i2 * win[2] + i3 * win[3];
        float hr = cr[v] + bhr, hz = cz[v] + bhz, hn = cn[v] + bhn_;
        float rr = sigm(gr + hr);
        float zz = sigm(gz + hz);
        float nn = tanhf(gn + rr * hn);
        float hold = sH[bl * Hd + c0];
        float hnew = (1.f - zz) * nn + zz * hold;
        sH[bl * Hd + c0] = hnew;
        sHh[bl * Hd + c0] = (_Float16)hnew;
      }
    }
    __syncthreads();

    // ---------- phase 3: gate-logit partials (4-way split) + theta ----------
    if (tid < BT * Mm * 4) {
      int g = tid >> 2, p = tid & 3;
      int bl = g / 3, m = g % 3;
      int c1 = p * 32, c2 = (p == 3) ? (Hd - 1) : (c1 + 32);
      float acc = 0.f;
      for (int c = c1; c < c2; ++c) acc += sH[bl * Hd + c] * sWd[c * Mm + m];
      sGLp[tid] = acc;
    } else if (tid < BT * Mm * 4 + BT) {
      int bl = tid - BT * Mm * 4;
      sTheta[bl] = sigm(sH[bl * Hd + (Hd - 1)]);
    }
    __syncthreads();

    // ---------- phase 4a: combine, softmax, theta blend, penalty ----------
    if (tid < BT) {
      float gl[3];
#pragma unroll
      for (int m = 0; m < 3; ++m) {
        int base = (tid * 3 + m) * 4;
        gl[m] = sbd[m] + sGLp[base] + sGLp[base + 1] + sGLp[base + 2] + sGLp[base + 3];
      }
      float mx = fmaxf(gl[0], fmaxf(gl[1], gl[2]));
      float e0 = __expf(gl[0] - mx), e1 = __expf(gl[1] - mx), e2 = __expf(gl[2] - mx);
      float inv = 1.f / (e0 + e1 + e2);
      float th = sTheta[tid];
      float p0 = sGate[tid * 3 + 0], p1 = sGate[tid * 3 + 1], p2 = sGate[tid * 3 + 2];
      float n0 = e0 * inv * th + p0 * (1.f - th);
      float n1 = e1 * inv * th + p1 * (1.f - th);
      float n2 = e2 * inv * th + p2 * (1.f - th);
      sGate[tid * 3 + 0] = n0; sGate[tid * 3 + 1] = n1; sGate[tid * 3 + 2] = n2;
      float pen = n0 * (1.f - n0) + n1 * (1.f - n1) + n2 * (1.f - n2);
      out[(size_t)Bt * Tt + (size_t)(b0 + tid) * Tt + t] = pen;
    }
    __syncthreads();

    // ---------- phase 4b: s_new, pred, err ----------
    for (int i = tid; i < BT * NSd; i += 256) {
      int bl = i >> 6, n = i & 63;
      float g0 = sGate[bl * 3 + 0], g1 = sGate[bl * 3 + 1], g2 = sGate[bl * 3 + 2];
      float v = g0 * sNSt[(bl * 3 + 0) * NSd + n] +
                g1 * sNSt[(bl * 3 + 1) * NSd + n] +
                g2 * sNSt[(bl * 3 + 2) * NSd + n];
      sSt[i] = v;
      sSh[i] = (_Float16)v;
      if (n == NSd - 1) {
        out[(size_t)(b0 + bl) * Tt + t] = v;   // pred
        sErr[bl] = v - sy_[bl];
      }
    }
    __syncthreads();
  }
}

extern "C" void kernel_launch(void* const* d_in, const int* in_sizes, int n_in,
                              void* d_out, int out_size, void* d_ws, size_t ws_size,
                              hipStream_t stream) {
  (void)in_sizes; (void)n_in; (void)out_size; (void)ws_size;
  const float* x   = (const float*)d_in[0];
  const float* y   = (const float*)d_in[1];
  const float* U   = (const float*)d_in[2];
  const float* S   = (const float*)d_in[3];
  const float* A   = (const float*)d_in[4];
  const float* Wih = (const float*)d_in[5];
  const float* Whh = (const float*)d_in[6];
  const float* bih = (const float*)d_in[7];
  const float* bhh = (const float*)d_in[8];
  const float* Wd  = (const float*)d_in[9];
  const float* bd  = (const float*)d_in[10];
  _Float16* whhF = (_Float16*)d_ws;  // 96 KB of fragment-ordered f16 W_hh

  disc_prep_whh<<<192, 256, 0, stream>>>(Whh, whhF);
  disc_main<<<16, 256, 0, stream>>>(x, y, U, S, A, Wih, bih, bhh, Wd, bd, whhF,
                                    (float*)d_out);
}
